// EncoderBlock_9088150798451
// MI455X (gfx1250) — matrix operand order repaired and checked by reading, not compiled, not run on hardware
//
#include <hip/hip_runtime.h>
#include <math.h>

// ---------------------------------------------------------------------------
// Problem constants (from reference)
// ---------------------------------------------------------------------------
#define BB    16
#define NN    576
#define CC    1024
#define HH    16
#define HD    64
#define HID   4096
#define LCUR  256
#define PREFIX 64
#define KEYS  832        // LCUR + NN
#define MROWS (BB*NN)    // 9216

// Workspace layout (bytes, 256-aligned by construction)
#define OFF_WQKV  0ull
#define OFF_WPROJ 6291456ull
#define OFF_WFC1  8388608ull
#define OFF_WFC2  16777216ull
#define OFF_H     25165824ull   // LN output (reused for LN2)
#define OFF_X1    44040192ull   // fp32 residual stream after attention
#define OFF_Q     81788928ull
#define OFF_K     100663296ull
#define OFF_V     127926272ull
#define OFF_O     155189248ull
#define OFF_MID   81788928ull   // MLP hidden, overlaps q/k/v/obuf (dead by then)

typedef __attribute__((ext_vector_type(16))) __bf16 v16bf;
typedef __attribute__((ext_vector_type(8)))  float  v8f;

// ---------------------------------------------------------------------------
// WMMA fragment loaders (16x16x32 bf16, wave32; this code's k order, same for A and B)
// A 16x32 (MxK), row-major source with leading dim ld:
//   lane (l&15) = row M; half (l>>4): K[h*16 .. h*16+15] contiguous
// B 32x16 (KxN), source is column-major (i.e. W^T row-major, ld = K stride):
//   lane (l&15) = col N; half (l>>4): K[h*16 .. h*16+15] contiguous
// ---------------------------------------------------------------------------
__device__ inline v16bf load_a16x32(const __bf16* __restrict__ p, int ld) {
  int lane = threadIdx.x & 31;
  int row = lane & 15, kh = lane >> 4;
  const __bf16* r0 = p + (size_t)row * ld + kh * 16;
  const __bf16* r1 = r0 + 8;
  v16bf a;
#pragma unroll
  for (int i = 0; i < 8; ++i) { a[i] = r0[i]; a[8 + i] = r1[i]; }
  return a;
}

__device__ inline v16bf load_b32x16(const __bf16* __restrict__ p, int ld) {
  int lane = threadIdx.x & 31;
  int col = lane & 15, kb = (lane >> 4) * 16;
  const __bf16* r = p + (size_t)col * ld + kb;
  v16bf b;
#pragma unroll
  for (int i = 0; i < 16; ++i) b[i] = r[i];
  return b;
}

__device__ inline v8f wmma_bf16(v16bf a, v16bf b, v8f c) {
  return __builtin_amdgcn_wmma_f32_16x16x32_bf16(false, a, false, b, (short)0, c,
                                                 false, false);
}

__device__ inline float redmax16(float x) {
#pragma unroll
  for (int m = 1; m < 16; m <<= 1) x = fmaxf(x, __shfl_xor(x, m, 32));
  return x;
}
__device__ inline float redsum16(float x) {
#pragma unroll
  for (int m = 1; m < 16; m <<= 1) x += __shfl_xor(x, m, 32);
  return x;
}

// ---------------------------------------------------------------------------
// Weight fp32 [R,C] -> bf16 transposed [C,R]
// ---------------------------------------------------------------------------
__global__ void convT_kernel(const float* __restrict__ in, __bf16* __restrict__ out,
                             int R, int C) {
  size_t idx = (size_t)blockIdx.x * blockDim.x + threadIdx.x;
  size_t total = (size_t)R * C;
  if (idx >= total) return;
  int r = (int)(idx / C), c = (int)(idx % C);
  out[(size_t)c * R + r] = (__bf16)in[idx];
}

// ---------------------------------------------------------------------------
// LayerNorm over C=1024 -> bf16
// ---------------------------------------------------------------------------
__global__ __launch_bounds__(256)
void ln_kernel(const float* __restrict__ x, const float* __restrict__ w,
               const float* __restrict__ b, __bf16* __restrict__ out) {
  int row = blockIdx.x, tid = threadIdx.x;
  const float* xr = x + (size_t)row * CC;
  __shared__ float red[256];
  float v[4], s = 0.f;
#pragma unroll
  for (int i = 0; i < 4; ++i) { v[i] = xr[tid + 256 * i]; s += v[i]; }
  red[tid] = s; __syncthreads();
  for (int st = 128; st > 0; st >>= 1) { if (tid < st) red[tid] += red[tid + st]; __syncthreads(); }
  float mean = red[0] * (1.f / CC);
  __syncthreads();
  s = 0.f;
#pragma unroll
  for (int i = 0; i < 4; ++i) { float d = v[i] - mean; s += d * d; }
  red[tid] = s; __syncthreads();
  for (int st = 128; st > 0; st >>= 1) { if (tid < st) red[tid] += red[tid + st]; __syncthreads(); }
  float rstd = rsqrtf(red[0] * (1.f / CC) + 1e-5f);
#pragma unroll
  for (int i = 0; i < 4; ++i) {
    int c = tid + 256 * i;
    out[(size_t)row * CC + c] = (__bf16)((v[i] - mean) * rstd * w[c] + b[c]);
  }
}

// ---------------------------------------------------------------------------
// Copy cache_k/v[:, :, 64:320, :] into effective key/value buffers (bf16)
// keff: [BH, KEYS, HD] row-major; veff: [BH, HD, KEYS] (transposed)
// ---------------------------------------------------------------------------
__global__ void cachefill_kernel(const float* __restrict__ ck, const float* __restrict__ cv,
                                 __bf16* __restrict__ keff, __bf16* __restrict__ veff) {
  size_t idx = (size_t)blockIdx.x * blockDim.x + threadIdx.x;  // BH*256*64
  int hd = (int)(idx & 63);
  int p  = (int)((idx >> 6) & 255);
  int bh = (int)(idx >> 14);
  if (bh >= BB * HH) return;
  int pos = PREFIX + p;
  size_t src = ((size_t)bh * 1024 + pos) * HD + hd;
  keff[((size_t)bh * KEYS + pos) * HD + hd] = (__bf16)ck[src];
  veff[((size_t)bh * HD + hd) * KEYS + pos] = (__bf16)cv[src];
}

// ---------------------------------------------------------------------------
// Generic bf16 WMMA GEMM: out[M,N] = A[M,K] @ WT[N,K]^T + bias, epilogues:
//   EPI 0: QKV scatter (q scaled, k/v into cache-concat layout)
//   EPI 1: proj: fp32 out = res + val
//   EPI 2: fc1: bf16 out = gelu(val)
//   EPI 3: fc2: fp32 out = res + val
// WG = 256 threads (8 waves) arranged 2(M) x 4(N): block tile 64 x 256.
// Each wave computes 32(M) x 64(N): 2 A-frags + 4 B-frags -> 8 WMMA per K-step.
// ---------------------------------------------------------------------------
template <int EPI>
__global__ __launch_bounds__(256)
void gemm_bf16_epi(const __bf16* __restrict__ A, const __bf16* __restrict__ WT,
                   const float* __restrict__ bias, const float* __restrict__ res,
                   float* __restrict__ outf, __bf16* __restrict__ outb,
                   __bf16* __restrict__ qb, __bf16* __restrict__ kb,
                   __bf16* __restrict__ vb, int M, int N, int K) {
  int w = threadIdx.x >> 5;
  int m0 = blockIdx.x * 64 + (w & 1) * 32;
  int n0 = blockIdx.y * 256 + (w >> 1) * 64;
  v8f zero = {0.f, 0.f, 0.f, 0.f, 0.f, 0.f, 0.f, 0.f};
  v8f c[2][4] = {{zero, zero, zero, zero}, {zero, zero, zero, zero}};
  const __bf16* arow0 = A + (size_t)m0 * K;
  const __bf16* arow1 = A + (size_t)(m0 + 16) * K;
  for (int k0 = 0; k0 < K; k0 += 32) {
    v16bf a0 = load_a16x32(arow0 + k0, K);
    v16bf a1 = load_a16x32(arow1 + k0, K);
#pragma unroll
    for (int n = 0; n < 4; ++n) {
      v16bf bf = load_b32x16(WT + (size_t)(n0 + n * 16) * K + k0, K);
      c[0][n] = wmma_bf16(a0, bf, c[0][n]);
      c[1][n] = wmma_bf16(a1, bf, c[1][n]);
    }
  }
  int lane = threadIdx.x & 31, lo = lane & 15, hi = lane >> 4;
#pragma unroll
  for (int mi = 0; mi < 2; ++mi) {
#pragma unroll
    for (int n = 0; n < 4; ++n) {
#pragma unroll
      for (int j = 0; j < 8; ++j) {
        int row = m0 + mi * 16 + j + 8 * hi;
        int col = n0 + n * 16 + lo;
        float val = c[mi][n][j] + bias[col];
        if constexpr (EPI == 0) {
          int bbi = row / NN, tok = row % NN;
          int part = col >> 10, c1 = col & 1023;
          int h = c1 >> 6, hd = c1 & 63;
          int bh = bbi * HH + h;
          if (part == 0) {
            qb[((size_t)bh * NN + tok) * HD + hd] = (__bf16)(val * 0.125f);
          } else {
            int pos = (tok < PREFIX) ? tok : tok + LCUR;
            if (part == 1) kb[((size_t)bh * KEYS + pos) * HD + hd] = (__bf16)val;
            else           vb[((size_t)bh * HD + hd) * KEYS + pos] = (__bf16)val;
          }
        } else if constexpr (EPI == 1) {
          size_t o = (size_t)row * N + col;
          outf[o] = res[o] + val;
        } else if constexpr (EPI == 2) {
          float g = 0.5f * val * (1.f + erff(val * 0.70710678118654752f));
          outb[(size_t)row * N + col] = (__bf16)g;
        } else {
          size_t o = (size_t)row * N + col;
          outf[o] = res[o] + val;
        }
      }
    }
  }
}

// ---------------------------------------------------------------------------
// Flash attention: grid = B*H*(N/64), block = 128 (4 waves, 16 queries each)
// q: [BH, N, HD] (pre-scaled), keff: [BH, KEYS, HD], veff: [BH, HD, KEYS]
// obuf: [B, N, C] bf16 (head-concat layout ready for proj GEMM)
// ---------------------------------------------------------------------------
__global__ __launch_bounds__(128)
void attn_kernel(const __bf16* __restrict__ q, const __bf16* __restrict__ keff,
                 const __bf16* __restrict__ veff, __bf16* __restrict__ obuf) {
  int w = threadIdx.x >> 5, lane = threadIdx.x & 31;
  int lo = lane & 15, hi = lane >> 4;
  int qt = blockIdx.x % 9;
  int bh = blockIdx.x / 9;
  int h = bh & 15, b = bh >> 4;
  int tok0 = qt * 64 + w * 16;

  const __bf16* qp = q + ((size_t)bh * NN + tok0) * HD;
  const __bf16* kp = keff + (size_t)bh * KEYS * HD;
  const __bf16* vp = veff + (size_t)bh * HD * KEYS;

  v16bf qa0 = load_a16x32(qp, HD);
  v16bf qa1 = load_a16x32(qp + 32, HD);

  __shared__ __bf16 plds[4][16][64];

  v8f zero = {0.f, 0.f, 0.f, 0.f, 0.f, 0.f, 0.f, 0.f};
  v8f oc[4] = {zero, zero, zero, zero};
  float m[8], l[8];
#pragma unroll
  for (int j = 0; j < 8; ++j) { m[j] = -3.0e38f; l[j] = 0.f; }

  for (int kb = 0; kb < KEYS / 64; ++kb) {
    int key0 = kb * 64;
    // ---- scores S = Q @ K^T (16 x 64) ----
    v8f s[4] = {zero, zero, zero, zero};
#pragma unroll
    for (int n = 0; n < 4; ++n) {
      v16bf b0 = load_b32x16(kp + (size_t)(key0 + n * 16) * HD, HD);
      s[n] = wmma_bf16(qa0, b0, s[n]);
      v16bf b1 = load_b32x16(kp + (size_t)(key0 + n * 16) * HD + 32, HD);
      s[n] = wmma_bf16(qa1, b1, s[n]);
    }
    // ---- online softmax (rows mapped j + 8*hi, cols across 16 lanes) ----
    float rs[8];
#pragma unroll
    for (int j = 0; j < 8; ++j) {
      float bm = fmaxf(fmaxf(s[0][j], s[1][j]), fmaxf(s[2][j], s[3][j]));
      bm = redmax16(bm);
      float mn = fmaxf(m[j], bm);
      float corr = __expf(m[j] - mn);
      m[j] = mn;
      l[j] *= corr;
#pragma unroll
      for (int n = 0; n < 4; ++n) oc[n][j] *= corr;
      rs[j] = 0.f;
    }
#pragma unroll
    for (int n = 0; n < 4; ++n)
#pragma unroll
      for (int j = 0; j < 8; ++j) {
        float p = __expf(s[n][j] - m[j]);
        s[n][j] = p;
        rs[j] += p;
      }
#pragma unroll
    for (int j = 0; j < 8; ++j) l[j] += redsum16(rs[j]);

    // ---- bounce P through LDS to get A-fragment layout ----
    __syncthreads();
#pragma unroll
    for (int n = 0; n < 4; ++n)
#pragma unroll
      for (int j = 0; j < 8; ++j)
        plds[w][j + 8 * hi][n * 16 + lo] = (__bf16)s[n][j];
    __syncthreads();
    const __bf16* pw = &plds[w][0][0];
    v16bf pa0 = load_a16x32(pw, 64);
    v16bf pa1 = load_a16x32(pw + 32, 64);

    // ---- O += P @ V ----
#pragma unroll
    for (int n = 0; n < 4; ++n) {
      v16bf v0 = load_b32x16(vp + (size_t)(n * 16) * KEYS + key0, KEYS);
      oc[n] = wmma_bf16(pa0, v0, oc[n]);
      v16bf v1 = load_b32x16(vp + (size_t)(n * 16) * KEYS + key0 + 32, KEYS);
      oc[n] = wmma_bf16(pa1, v1, oc[n]);
    }
  }

  // ---- normalize and store head-concat output ----
#pragma unroll
  for (int n = 0; n < 4; ++n)
#pragma unroll
    for (int j = 0; j < 8; ++j) {
      int tok = tok0 + j + 8 * hi;
      int col = h * HD + n * 16 + lo;
      obuf[((size_t)b * NN + tok) * CC + col] = (__bf16)(oc[n][j] / l[j]);
    }
}

// ---------------------------------------------------------------------------
// Host launch
// ---------------------------------------------------------------------------
extern "C" void kernel_launch(void* const* d_in, const int* in_sizes, int n_in,
                              void* d_out, int out_size, void* d_ws, size_t ws_size,
                              hipStream_t stream) {
  const float* x       = (const float*)d_in[0];
  const float* cache_k = (const float*)d_in[1];
  const float* cache_v = (const float*)d_in[2];
  const float* qkv_w   = (const float*)d_in[3];
  const float* qkv_b   = (const float*)d_in[4];
  const float* proj_w  = (const float*)d_in[5];
  const float* proj_b  = (const float*)d_in[6];
  const float* n1w     = (const float*)d_in[7];
  const float* n1b     = (const float*)d_in[8];
  const float* n2w     = (const float*)d_in[9];
  const float* n2b     = (const float*)d_in[10];
  const float* fc1_w   = (const float*)d_in[11];
  const float* fc1_b   = (const float*)d_in[12];
  const float* fc2_w   = (const float*)d_in[13];
  const float* fc2_b   = (const float*)d_in[14];
  float* out = (float*)d_out;

  char* ws = (char*)d_ws;
  __bf16* wqkvT = (__bf16*)(ws + OFF_WQKV);
  __bf16* wprojT= (__bf16*)(ws + OFF_WPROJ);
  __bf16* wfc1T = (__bf16*)(ws + OFF_WFC1);
  __bf16* wfc2T = (__bf16*)(ws + OFF_WFC2);
  __bf16* hbuf  = (__bf16*)(ws + OFF_H);
  float*  x1    = (float*)(ws + OFF_X1);
  __bf16* qbuf  = (__bf16*)(ws + OFF_Q);
  __bf16* keff  = (__bf16*)(ws + OFF_K);
  __bf16* veff  = (__bf16*)(ws + OFF_V);
  __bf16* obuf  = (__bf16*)(ws + OFF_O);
  __bf16* mid   = (__bf16*)(ws + OFF_MID);

  // weights -> bf16 transposed
  convT_kernel<<<(CC * 3 * CC) / 256, 256, 0, stream>>>(qkv_w, wqkvT, CC, 3 * CC);
  convT_kernel<<<(CC * CC) / 256, 256, 0, stream>>>(proj_w, wprojT, CC, CC);
  convT_kernel<<<(CC * HID) / 256, 256, 0, stream>>>(fc1_w, wfc1T, CC, HID);
  convT_kernel<<<(HID * CC) / 256, 256, 0, stream>>>(fc2_w, wfc2T, HID, CC);

  // LN1 + cache region of keys/values
  ln_kernel<<<MROWS, 256, 0, stream>>>(x, n1w, n1b, hbuf);
  cachefill_kernel<<<(BB * HH * 256 * 64) / 256, 256, 0, stream>>>(cache_k, cache_v,
                                                                   keff, veff);
  // QKV GEMM + scatter  (block tile 64 x 256)
  gemm_bf16_epi<0><<<dim3(MROWS / 64, (3 * CC) / 256), 256, 0, stream>>>(
      hbuf, wqkvT, qkv_b, nullptr, nullptr, nullptr, qbuf, keff, veff,
      MROWS, 3 * CC, CC);
  // attention
  attn_kernel<<<BB * HH * (NN / 64), 128, 0, stream>>>(qbuf, keff, veff, obuf);
  // proj + residual -> x1 (fp32)
  gemm_bf16_epi<1><<<dim3(MROWS / 64, CC / 256), 256, 0, stream>>>(
      obuf, wprojT, proj_b, x, x1, nullptr, nullptr, nullptr, nullptr,
      MROWS, CC, CC);
  // LN2 -> hbuf (reuse)
  ln_kernel<<<MROWS, 256, 0, stream>>>(x1, n2w, n2b, hbuf);
  // FC1 + GELU -> mid
  gemm_bf16_epi<2><<<dim3(MROWS / 64, HID / 256), 256, 0, stream>>>(
      hbuf, wfc1T, fc1_b, nullptr, nullptr, mid, nullptr, nullptr, nullptr,
      MROWS, HID, CC);
  // FC2 + residual -> out
  gemm_bf16_epi<3><<<dim3(MROWS / 64, CC / 256), 256, 0, stream>>>(
      mid, wfc2T, fc2_b, x1, out, nullptr, nullptr, nullptr, nullptr,
      MROWS, CC, HID);
}
